// AttentionBlock_70763881169154
// MI455X (gfx1250) — matrix operand order repaired and checked by reading, not compiled, not run on hardware
//
#include <hip/hip_runtime.h>

typedef __attribute__((ext_vector_type(16))) _Float16 v16h;
typedef __attribute__((ext_vector_type(8)))  _Float16 v8h;
typedef __attribute__((ext_vector_type(8)))  float    v8f;

#define BATCH 4
#define CHN   64
#define NPOS  4096     // 64*64 spatial positions
#define TOK   77
#define TOKP  96       // padded token count (3 tiles of 32)
#define EPSV  1e-5f
#define ATTN_SCALE 0.125f

// ---------------------------------------------------------------------------
// WMMA helper: D = A(16x32 f16) * B(32x16 f16) + C(16x16 f32)
// ---------------------------------------------------------------------------
__device__ __forceinline__ v8f wmma_f16(v16h a, v16h b, v8f c) {
  return __builtin_amdgcn_wmma_f32_16x16x32_f16(false, a, false, b,
                                                (short)0, c, false, false);
}

// ---------------------------------------------------------------------------
// CDNA5 async global->LDS copy (ASYNCcnt-tracked, no VGPR round trip).
// lds_off = wave-relative LDS byte offset (low 32 bits of generic pointer),
// g       = 16-byte-aligned global source.
// ---------------------------------------------------------------------------
__device__ __forceinline__ void async_b128(unsigned lds_off, const _Float16* g) {
  asm volatile("global_load_async_to_lds_b128 %0, %1, off"
               :: "v"(lds_off), "v"(g) : "memory");
}
__device__ __forceinline__ void wait_async0() {
  asm volatile("s_wait_asynccnt 0x0" ::: "memory");
}

// ---------------------------------------------------------------------------
// 1) GroupNorm statistics: one block per (batch, group). Each group = 2
//    contiguous channels = 8192 contiguous floats.
// ---------------------------------------------------------------------------
__global__ __launch_bounds__(256) void gn_stats_k(const float* __restrict__ x,
                                                  float* __restrict__ stats) {
  int bg = blockIdx.x;                       // b*32 + g, 0..127
  const float* p = x + (size_t)bg * 8192;    // (b*64 + 2g)*4096 == bg*8192
  float s = 0.f, sq = 0.f;
  for (int i = threadIdx.x; i < 8192; i += 256) {
    float v = p[i]; s += v; sq += v * v;
  }
  __shared__ float rs[256], rq[256];
  rs[threadIdx.x] = s; rq[threadIdx.x] = sq;
  __syncthreads();
  for (int off = 128; off > 0; off >>= 1) {
    if (threadIdx.x < off) {
      rs[threadIdx.x] += rs[threadIdx.x + off];
      rq[threadIdx.x] += rq[threadIdx.x + off];
    }
    __syncthreads();
  }
  if (threadIdx.x == 0) {
    float mu  = rs[0] * (1.f / 8192.f);
    float var = rq[0] * (1.f / 8192.f) - mu * mu;
    stats[bg * 2 + 0] = mu;
    stats[bg * 2 + 1] = rsqrtf(var + EPSV);
  }
}

// ---------------------------------------------------------------------------
// 2) Apply GroupNorm + QKV 1x1 conv. q/k written (B,N,C) f16;
//    v written CHANNEL-MAJOR (B,C,N) f16 so attention tiles are contiguous
//    rows for the async-to-LDS copy (async copies cannot transpose).
// ---------------------------------------------------------------------------
__global__ __launch_bounds__(256) void gn_qkv_k(
    const float* __restrict__ x, const float* __restrict__ stats,
    const float* __restrict__ gsc, const float* __restrict__ gbi,
    const float* __restrict__ qkvw, const float* __restrict__ qkvb,
    _Float16* __restrict__ q, _Float16* __restrict__ k, _Float16* __restrict__ v) {
  int b  = blockIdx.x >> 6;
  int p0 = (blockIdx.x & 63) * 64;
  __shared__ float hn[64 * 64];              // [channel][local position]
  for (int i = threadIdx.x; i < 4096; i += 256) {
    int c = i >> 6, p = i & 63;
    int g = c >> 1;
    float mu   = stats[(b * 32 + g) * 2 + 0];
    float rstd = stats[(b * 32 + g) * 2 + 1];
    float val  = x[((size_t)(b * 64 + c)) * 4096 + p0 + p];
    hn[i] = (val - mu) * rstd * gsc[c] + gbi[c];
  }
  __syncthreads();
  for (int i = threadIdx.x; i < 64 * 192; i += 256) {
    int p = i / 192, o = i - p * 192;
    float acc = qkvb[o];
    const float* wr = qkvw + o * 64;
    for (int c = 0; c < 64; ++c) acc += hn[c * 64 + p] * wr[c];
    int oc = o & 63;
    int n  = p0 + p;
    if (o < 64)       q[((size_t)(b * NPOS + n)) * 64 + oc] = (_Float16)acc;
    else if (o < 128) k[((size_t)(b * NPOS + n)) * 64 + oc] = (_Float16)acc;
    else              v[((size_t)(b * 64 + oc)) * 4096 + n] = (_Float16)acc;
  }
}

// ---------------------------------------------------------------------------
// 3) Self-attention, flash style. Wave = one 16-row Q tile; block (8 waves)
//    shares 32-key K/V LDS tiles, double-buffered via async global->LDS
//    copies so tile t+1 streams in while tile t's 8 WMMAs execute.
//    Logits are tiny (~N(0,0.03)): plain exp + deferred normalization.
// ---------------------------------------------------------------------------
__global__ __launch_bounds__(256) void self_attn_k(
    const _Float16* __restrict__ q, const _Float16* __restrict__ k,
    const _Float16* __restrict__ vT, float* __restrict__ out) {
  __shared__ __attribute__((aligned(32))) _Float16 kt[2][32 * 64];   // [key][ch]
  __shared__ __attribute__((aligned(32))) _Float16 vtT[2][64 * 32];  // [ch][key]
  __shared__ __attribute__((aligned(32))) _Float16 pst[8][16 * 32];

  int tid = threadIdx.x;
  int wv = tid >> 5, lane = tid & 31;
  int m = lane & 15, hi = lane >> 4;

  int b     = blockIdx.x >> 5;                 // 32 blocks per batch
  int qtile = (blockIdx.x << 3) + wv;          // 0..1023
  int row0  = (qtile & 255) << 4;              // query row within batch
  size_t base = (size_t)b * NPOS * 64;

  // per-thread async copy slices (fixed across tiles)
  int ki8  = tid * 8;                          // K tile: contiguous copy
  int vrow = tid >> 2, vseg = (tid & 3) * 8;   // V tile: row=channel
  const _Float16* vrowg = vT + (size_t)(b * 64 + vrow) * 4096 + vseg;

  // Q tile -> two A operands (channels 0..31 and 32..63), contiguous K order:
  // lanes 0-15 hold K 0..15; lanes 16-31 hold K 16..31.
  const _Float16* qrow = q + base + (size_t)(row0 + m) * 64;
  v8h q00 = *(const v8h*)(qrow +  0 + hi * 16);
  v8h q01 = *(const v8h*)(qrow +  8 + hi * 16);
  v8h q10 = *(const v8h*)(qrow + 32 + hi * 16);
  v8h q11 = *(const v8h*)(qrow + 40 + hi * 16);
  v16h a0, a1;
#pragma unroll
  for (int i = 0; i < 8; ++i) {
    a0[i] = q00[i]; a0[i + 8] = q01[i];
    a1[i] = q10[i]; a1[i + 8] = q11[i];
  }

  v8f o0 = {0,0,0,0,0,0,0,0}, o1 = {0,0,0,0,0,0,0,0};
  v8f o2 = {0,0,0,0,0,0,0,0}, o3 = {0,0,0,0,0,0,0,0};
  v8f rsum = {0,0,0,0,0,0,0,0};

  // prologue: stream tile 0 into buffer 0
  async_b128((unsigned)(size_t)&kt[0][ki8], k + base + ki8);
  async_b128((unsigned)(size_t)&vtT[0][vrow * 32 + vseg], vrowg);

  for (int ktile = 0; ktile < 128; ++ktile) {
    int cur = ktile & 1;
    wait_async0();                // this thread's slices of tile `ktile` landed
    __syncthreads();              // ...and everyone else's
    if (ktile + 1 < 128) {        // prefetch next tile into the other buffer
      int kb = (ktile + 1) * 32;
      async_b128((unsigned)(size_t)&kt[cur ^ 1][ki8],
                 k + base + (size_t)kb * 64 + ki8);
      async_b128((unsigned)(size_t)&vtT[cur ^ 1][vrow * 32 + vseg], vrowg + kb);
    }

    // S = Q * K^T for two 16-key column tiles, then P = exp(S*scale)
#pragma unroll
    for (int ct = 0; ct < 2; ++ct) {
      int key = ct * 16 + m;
      v16h b0 = *(const v16h*)(&kt[cur][key * 64 +  0 + hi * 16]);
      v16h b1 = *(const v16h*)(&kt[cur][key * 64 + 32 + hi * 16]);
      v8f s = {0,0,0,0,0,0,0,0};
      s = wmma_f16(a0, b0, s);
      s = wmma_f16(a1, b1, s);
#pragma unroll
      for (int r = 0; r < 8; ++r) {
        float pv = __expf(s[r] * ATTN_SCALE);
        rsum[r] += pv;
        pst[wv][(r + 8 * hi) * 32 + ct * 16 + m] = (_Float16)pv;
      }
    }

    // P back as A operand (per-wave LDS; DS ops are in-order within a wave)
    v8h p0 = *(const v8h*)(&pst[wv][m * 32 +  0 + hi * 16]);
    v8h p1 = *(const v8h*)(&pst[wv][m * 32 +  8 + hi * 16]);
    v16h pa;
#pragma unroll
    for (int i = 0; i < 8; ++i) { pa[i] = p0[i]; pa[i + 8] = p1[i]; }

    // O += P * V over four 16-channel column tiles
    v16h bv0 = *(const v16h*)(&vtT[cur][( 0 + m) * 32 + hi * 16]);
    v16h bv1 = *(const v16h*)(&vtT[cur][(16 + m) * 32 + hi * 16]);
    v16h bv2 = *(const v16h*)(&vtT[cur][(32 + m) * 32 + hi * 16]);
    v16h bv3 = *(const v16h*)(&vtT[cur][(48 + m) * 32 + hi * 16]);
    o0 = wmma_f16(pa, bv0, o0);
    o1 = wmma_f16(pa, bv1, o1);
    o2 = wmma_f16(pa, bv2, o2);
    o3 = wmma_f16(pa, bv3, o3);
    __syncthreads();              // all waves done with buffer `cur`
  }

  // finish softmax denominator: reduce across the 16 lanes of each half-wave
#pragma unroll
  for (int r = 0; r < 8; ++r) {
    float t = rsum[r];
    t += __shfl_xor(t, 1, 32);
    t += __shfl_xor(t, 2, 32);
    t += __shfl_xor(t, 4, 32);
    t += __shfl_xor(t, 8, 32);
    rsum[r] = t;
  }

  float* orow = out + base;
#pragma unroll
  for (int r = 0; r < 8; ++r) {
    float inv = 1.0f / rsum[r];
    size_t rowg = (size_t)(row0 + r + 8 * hi) * 64;
    orow[rowg +  0 + m] = o0[r] * inv;
    orow[rowg + 16 + m] = o1[r] * inv;
    orow[rowg + 32 + m] = o2[r] * inv;
    orow[rowg + 48 + m] = o3[r] * inv;
  }
}

// ---------------------------------------------------------------------------
// 4) proj 1x1 conv + residual: h2 = x + proj_w @ self_out + proj_b
// ---------------------------------------------------------------------------
__global__ __launch_bounds__(256) void proj_res_k(
    const float* __restrict__ x, const float* __restrict__ so,
    const float* __restrict__ pw, const float* __restrict__ pb,
    float* __restrict__ h2) {
  int b  = blockIdx.x >> 6;
  int p0 = (blockIdx.x & 63) * 64;
  __shared__ float st[4096];                 // [position][channel]
  size_t sbase = (size_t)b * NPOS * 64 + (size_t)p0 * 64;
  for (int i = threadIdx.x; i < 4096; i += 256) st[i] = so[sbase + i];
  __syncthreads();
  for (int i = threadIdx.x; i < 4096; i += 256) {
    int o = i >> 6, p = i & 63;
    float acc = pb[o];
    const float* wr = pw + o * 64;
    const float* sp = st + p * 64;
    for (int c = 0; c < 64; ++c) acc += sp[c] * wr[c];
    size_t xi = ((size_t)(b * 64 + o)) * 4096 + p0 + p;
    h2[xi] = x[xi] + acc;
  }
}

// ---------------------------------------------------------------------------
// 5) LayerNorm over channels + tq projection (wq stored (in,out)).
// ---------------------------------------------------------------------------
__global__ __launch_bounds__(256) void ln_tq_k(
    const float* __restrict__ h2, const float* __restrict__ lsc,
    const float* __restrict__ lbi, const float* __restrict__ wq,
    const float* __restrict__ bq, _Float16* __restrict__ tq) {
  int gid = blockIdx.x * 256 + threadIdx.x;    // 0..16383
  int b = gid >> 12, n = gid & 4095;
  float tf[64];
  float s = 0.f;
  for (int c = 0; c < 64; ++c) {
    float val = h2[((size_t)(b * 64 + c)) * 4096 + n];
    tf[c] = val; s += val;
  }
  float mu = s * (1.f / 64.f);
  float vq = 0.f;
  for (int c = 0; c < 64; ++c) { float d = tf[c] - mu; vq += d * d; }
  float rstd = rsqrtf(vq * (1.f / 64.f) + EPSV);
  for (int c = 0; c < 64; ++c) tf[c] = (tf[c] - mu) * rstd * lsc[c] + lbi[c];
  _Float16* dst = tq + (size_t)gid * 64;
  for (int o = 0; o < 64; ++o) {
    float acc = bq[o];
    for (int c = 0; c < 64; ++c) acc += tf[c] * wq[c * 64 + o];
    dst[o] = (_Float16)acc;
  }
}

// ---------------------------------------------------------------------------
// 6) context -> tk/tv, padded 77 -> 96 tokens (zeros; masked later anyway).
//    tk row-major (B,96,C); tv CHANNEL-MAJOR (B,C,96) for async tile copies.
// ---------------------------------------------------------------------------
__global__ __launch_bounds__(64) void ctx_kv_k(
    const float* __restrict__ ctx, const float* __restrict__ wk,
    const float* __restrict__ bk, const float* __restrict__ wv,
    const float* __restrict__ bv,
    _Float16* __restrict__ tk, _Float16* __restrict__ tv) {
  int b = blockIdx.x / TOKP, t = blockIdx.x % TOKP, o = threadIdx.x;
  size_t ki = ((size_t)(b * TOKP + t)) * 64 + o;
  size_t vi = ((size_t)(b * 64 + o)) * TOKP + t;
  if (t >= TOK) { tk[ki] = (_Float16)0.f; tv[vi] = (_Float16)0.f; return; }
  const float* cr = ctx + ((size_t)(b * TOK + t)) * 768;
  float ka = bk[o], va = bv[o];
  for (int d = 0; d < 768; ++d) {
    float cv = cr[d];
    ka += cv * wk[d * 64 + o];
    va += cv * wv[d * 64 + o];
  }
  tk[ki] = (_Float16)ka; tv[vi] = (_Float16)va;
}

// ---------------------------------------------------------------------------
// 7) Cross-attention flash: 3 key tiles of 32 (96 padded tokens), columns
//    >= 77 masked to P=0. Async double-buffered K/V tiles like self-attn.
// ---------------------------------------------------------------------------
__global__ __launch_bounds__(256) void cross_attn_k(
    const _Float16* __restrict__ tq, const _Float16* __restrict__ tk,
    const _Float16* __restrict__ tvT, float* __restrict__ tout) {
  __shared__ __attribute__((aligned(32))) _Float16 kt[2][32 * 64];
  __shared__ __attribute__((aligned(32))) _Float16 vtT[2][64 * 32];
  __shared__ __attribute__((aligned(32))) _Float16 pst[8][16 * 32];

  int tid = threadIdx.x;
  int wv = tid >> 5, lane = tid & 31;
  int m = lane & 15, hi = lane >> 4;

  int b     = blockIdx.x >> 5;
  int qtile = (blockIdx.x << 3) + wv;
  int row0  = (qtile & 255) << 4;
  size_t qbase  = (size_t)b * NPOS * 64;
  size_t kvbase = (size_t)b * TOKP * 64;

  int ki8  = tid * 8;
  int vrow = tid >> 2, vseg = (tid & 3) * 8;
  const _Float16* vrowg = tvT + (size_t)(b * 64 + vrow) * TOKP + vseg;

  const _Float16* qrow = tq + qbase + (size_t)(row0 + m) * 64;
  v8h q00 = *(const v8h*)(qrow +  0 + hi * 16);
  v8h q01 = *(const v8h*)(qrow +  8 + hi * 16);
  v8h q10 = *(const v8h*)(qrow + 32 + hi * 16);
  v8h q11 = *(const v8h*)(qrow + 40 + hi * 16);
  v16h a0, a1;
#pragma unroll
  for (int i = 0; i < 8; ++i) {
    a0[i] = q00[i]; a0[i + 8] = q01[i];
    a1[i] = q10[i]; a1[i + 8] = q11[i];
  }

  v8f o0 = {0,0,0,0,0,0,0,0}, o1 = {0,0,0,0,0,0,0,0};
  v8f o2 = {0,0,0,0,0,0,0,0}, o3 = {0,0,0,0,0,0,0,0};
  v8f rsum = {0,0,0,0,0,0,0,0};

  async_b128((unsigned)(size_t)&kt[0][ki8], tk + kvbase + ki8);
  async_b128((unsigned)(size_t)&vtT[0][vrow * 32 + vseg], vrowg);

  for (int ktile = 0; ktile < 3; ++ktile) {
    int cur = ktile & 1;
    int kb  = ktile * 32;
    wait_async0();
    __syncthreads();
    if (ktile + 1 < 3) {
      int nb = (ktile + 1) * 32;
      async_b128((unsigned)(size_t)&kt[cur ^ 1][ki8],
                 tk + kvbase + (size_t)nb * 64 + ki8);
      async_b128((unsigned)(size_t)&vtT[cur ^ 1][vrow * 32 + vseg], vrowg + nb);
    }

#pragma unroll
    for (int ct = 0; ct < 2; ++ct) {
      int key = ct * 16 + m;
      v16h b0 = *(const v16h*)(&kt[cur][key * 64 +  0 + hi * 16]);
      v16h b1 = *(const v16h*)(&kt[cur][key * 64 + 32 + hi * 16]);
      v8f s = {0,0,0,0,0,0,0,0};
      s = wmma_f16(a0, b0, s);
      s = wmma_f16(a1, b1, s);
      bool valid = (kb + ct * 16 + m) < TOK;
#pragma unroll
      for (int r = 0; r < 8; ++r) {
        float pv = valid ? __expf(s[r] * ATTN_SCALE) : 0.f;
        rsum[r] += pv;
        pst[wv][(r + 8 * hi) * 32 + ct * 16 + m] = (_Float16)pv;
      }
    }

    v8h p0 = *(const v8h*)(&pst[wv][m * 32 +  0 + hi * 16]);
    v8h p1 = *(const v8h*)(&pst[wv][m * 32 +  8 + hi * 16]);
    v16h pa;
#pragma unroll
    for (int i = 0; i < 8; ++i) { pa[i] = p0[i]; pa[i + 8] = p1[i]; }

    v16h bv0 = *(const v16h*)(&vtT[cur][( 0 + m) * 32 + hi * 16]);
    v16h bv1 = *(const v16h*)(&vtT[cur][(16 + m) * 32 + hi * 16]);
    v16h bv2 = *(const v16h*)(&vtT[cur][(32 + m) * 32 + hi * 16]);
    v16h bv3 = *(const v16h*)(&vtT[cur][(48 + m) * 32 + hi * 16]);
    o0 = wmma_f16(pa, bv0, o0);
    o1 = wmma_f16(pa, bv1, o1);
    o2 = wmma_f16(pa, bv2, o2);
    o3 = wmma_f16(pa, bv3, o3);
    __syncthreads();
  }

#pragma unroll
  for (int r = 0; r < 8; ++r) {
    float t = rsum[r];
    t += __shfl_xor(t, 1, 32);
    t += __shfl_xor(t, 2, 32);
    t += __shfl_xor(t, 4, 32);
    t += __shfl_xor(t, 8, 32);
    rsum[r] = t;
  }

  float* orow = tout + qbase;
#pragma unroll
  for (int r = 0; r < 8; ++r) {
    float inv = 1.0f / rsum[r];
    size_t rowg = (size_t)(row0 + r + 8 * hi) * 64;
    orow[rowg +  0 + m] = o0[r] * inv;
    orow[rowg + 16 + m] = o1[r] * inv;
    orow[rowg + 32 + m] = o2[r] * inv;
    orow[rowg + 48 + m] = o3[r] * inv;
  }
}

// ---------------------------------------------------------------------------
// 8) Final: out[b,c,hh,w] = h2[b,c,hh,w] + sum_w2 tout[b, c*64+w2, hh]*wo[w2,w]
//           + bo[w]   (faithful transpose-scramble + Linear over W)
// ---------------------------------------------------------------------------
__global__ __launch_bounds__(64) void final_k(
    const float* __restrict__ h2, const float* __restrict__ tout,
    const float* __restrict__ wo, const float* __restrict__ bo,
    float* __restrict__ out) {
  int id = blockIdx.x;                       // B*C*H = 16384
  int b = id >> 12, c = (id >> 6) & 63, hh = id & 63;
  __shared__ float s[64];
  int w = threadIdx.x;
  s[w] = tout[(((size_t)b * NPOS) + c * 64 + w) * 64 + hh];
  __syncthreads();
  size_t oi = ((size_t)(b * 64 + c)) * 4096 + hh * 64 + w;
  float acc = bo[w] + h2[oi];
  for (int w2 = 0; w2 < 64; ++w2) acc += s[w2] * wo[w2 * 64 + w];
  out[oi] = acc;
}

// ---------------------------------------------------------------------------
extern "C" void kernel_launch(void* const* d_in, const int* in_sizes, int n_in,
                              void* d_out, int out_size, void* d_ws, size_t ws_size,
                              hipStream_t stream) {
  (void)in_sizes; (void)n_in; (void)out_size; (void)ws_size;
  const float* x    = (const float*)d_in[0];
  const float* ctx  = (const float*)d_in[1];
  const float* gsc  = (const float*)d_in[2];
  const float* gbi  = (const float*)d_in[3];
  const float* qkvw = (const float*)d_in[4];
  const float* qkvb = (const float*)d_in[5];
  const float* pw   = (const float*)d_in[6];
  const float* pb   = (const float*)d_in[7];
  const float* lsc  = (const float*)d_in[8];
  const float* lbi  = (const float*)d_in[9];
  const float* wq   = (const float*)d_in[10];
  const float* bq   = (const float*)d_in[11];
  const float* wk   = (const float*)d_in[12];
  const float* bk   = (const float*)d_in[13];
  const float* wv   = (const float*)d_in[14];
  const float* bv   = (const float*)d_in[15];
  const float* wo   = (const float*)d_in[16];
  const float* bo   = (const float*)d_in[17];

  char* ws = (char*)d_ws;
  size_t off = 0;
  const size_t QKV_HALF = (size_t)BATCH * NPOS * 64 * sizeof(_Float16); // 2 MB
  const size_t BNC_F32  = (size_t)BATCH * NPOS * 64 * sizeof(float);   // 4 MB
  const size_t KVP_HALF = (size_t)BATCH * TOKP * 64 * sizeof(_Float16);

  float*     stats = (float*)(ws + off);     off += 4096;
  _Float16*  q     = (_Float16*)(ws + off);  off += QKV_HALF;
  _Float16*  kk    = (_Float16*)(ws + off);  off += QKV_HALF;
  _Float16*  vvT   = (_Float16*)(ws + off);  off += QKV_HALF;   // (B,C,N)
  float*     so    = (float*)(ws + off);     off += BNC_F32;
  float*     h2    = (float*)(ws + off);     off += BNC_F32;
  _Float16*  tq    = (_Float16*)(ws + off);  off += QKV_HALF;
  _Float16*  tkp   = (_Float16*)(ws + off);  off += KVP_HALF;   // (B,96,C)
  _Float16*  tvpT  = (_Float16*)(ws + off);  off += KVP_HALF;   // (B,C,96)
  float*     tout  = (float*)(ws + off);     off += BNC_F32;

  gn_stats_k  <<<128,   256, 0, stream>>>(x, stats);
  gn_qkv_k    <<<256,   256, 0, stream>>>(x, stats, gsc, gbi, qkvw, qkvb, q, kk, vvT);
  self_attn_k <<<128,   256, 0, stream>>>(q, kk, vvT, so);
  proj_res_k  <<<256,   256, 0, stream>>>(x, so, pw, pb, h2);
  ln_tq_k     <<<64,    256, 0, stream>>>(h2, lsc, lbi, wq, bq, tq);
  ctx_kv_k    <<<BATCH * TOKP, 64, 0, stream>>>(ctx, wk, bk, wv, bv, tkp, tvpT);
  cross_attn_k<<<128,   256, 0, stream>>>(tq, tkp, tvpT, tout);
  final_k     <<<16384, 64,  0, stream>>>(h2, tout, wo, bo, (float*)d_out);
}
